// RelPositionMultiHeadAttention_8847632629855
// MI455X (gfx1250) — matrix operand order repaired and checked by reading, not compiled, not run on hardware
//
#include <hip/hip_runtime.h>
#include <hip/hip_bf16.h>

// ---------------- problem constants ----------------
#define BSZ    8
#define NSEQ   1024
#define MSEQ   1024
#define HHEADS 8
#define DHEAD  64
#define DMODEL 512

// LDS row strides (floats / bf16 elems) with padding to dodge bank conflicts
#define LGS  1028   // logits rows (float)
#define ATTS 1032   // attn rows (bf16)
#define WTS  40     // weight-tile row stride (bf16 elems, 32 + 8 pad)

typedef __attribute__((ext_vector_type(16))) __bf16 v16bf;
typedef __attribute__((ext_vector_type(8)))  float  v8f;

union AFrag { v16bf v; __bf16 e[16]; uint4 q[2]; };
union CFrag { v8f v; float f[8]; };

// native hardware convert path (v_cvt_pk_bf16_f32), RNE
static __device__ __forceinline__ unsigned short bfBits(float f) {
    __bf16 b = (__bf16)f;
    unsigned short s; __builtin_memcpy(&s, &b, 2);
    return s;
}
static __device__ __forceinline__ unsigned pkBf(float lo, float hi) {
    return (unsigned)bfBits(lo) | ((unsigned)bfBits(hi) << 16);
}
static __device__ __forceinline__ v8f wmma_bf16(v16bf a, v16bf b, v8f c) {
    // D(f32 16x16) = A(bf16 16x32) * B(bf16 32x16) + C
    return __builtin_amdgcn_wmma_f32_16x16x32_bf16(false, a, false, b,
                                                   (short)0, c, false, false);
}

// cooperative fill of a transposed 32x64 bf16 weight tile:
// wt[d][kk] = (bf16) src[(k0+kk)*srcStride + d0base + d]
// thread owns a 4(d) x 4(kk) sub-block: 4x global_load_b128, 8x ds_store_b32
static __device__ __forceinline__ void fill_wtile(
    unsigned short (*wt)[WTS], const float* __restrict__ src,
    size_t srcStride, int tid)
{
    const int d0  = (tid & 15) * 4;   // 0..60
    const int kk0 = (tid >> 4) * 4;   // 0..28
    const float4 w0 = *(const float4*)(src + (size_t)(kk0 + 0) * srcStride + d0);
    const float4 w1 = *(const float4*)(src + (size_t)(kk0 + 1) * srcStride + d0);
    const float4 w2 = *(const float4*)(src + (size_t)(kk0 + 2) * srcStride + d0);
    const float4 w3 = *(const float4*)(src + (size_t)(kk0 + 3) * srcStride + d0);
    *(unsigned*)&wt[d0 + 0][kk0 + 0] = pkBf(w0.x, w1.x);
    *(unsigned*)&wt[d0 + 0][kk0 + 2] = pkBf(w2.x, w3.x);
    *(unsigned*)&wt[d0 + 1][kk0 + 0] = pkBf(w0.y, w1.y);
    *(unsigned*)&wt[d0 + 1][kk0 + 2] = pkBf(w2.y, w3.y);
    *(unsigned*)&wt[d0 + 2][kk0 + 0] = pkBf(w0.z, w1.z);
    *(unsigned*)&wt[d0 + 2][kk0 + 2] = pkBf(w2.z, w3.z);
    *(unsigned*)&wt[d0 + 3][kk0 + 0] = pkBf(w0.w, w1.w);
    *(unsigned*)&wt[d0 + 3][kk0 + 2] = pkBf(w2.w, w3.w);
}

// =====================================================================
// Kernel 1: per-head projections q(+biases), k, v(transposed), p  -> bf16
// Block = 4 waves sharing the same (mat, b, h) -> same weight slice W[h].
// Each k-step stages a transposed 32x64 bf16 weight tile in LDS once per
// block; each wave then computes a 16x64 tile: GEMM [16,512] x [512,64].
// mat: 0=q (writes q_u,q_v), 1=k, 2=v(->vt), 3=p
// =====================================================================
__global__ void __launch_bounds__(128) proj_kernel(
    const float* __restrict__ query, const float* __restrict__ key,
    const float* __restrict__ value, const float* __restrict__ pos,
    const float* __restrict__ qkern, const float* __restrict__ kkern,
    const float* __restrict__ vkern, const float* __restrict__ pkern,
    const float* __restrict__ bias_u, const float* __restrict__ bias_v,
    unsigned short* __restrict__ qu, unsigned short* __restrict__ qv,
    unsigned short* __restrict__ kws, unsigned short* __restrict__ pws,
    unsigned short* __restrict__ vt)
{
    __shared__ unsigned short wt[DHEAD][WTS];   // wt[d][kk]: transposed bf16 tile

    const int tid  = threadIdx.x;
    const int wid  = tid >> 5;
    const int lane = tid & 31;
    const int job  = blockIdx.x * 4 + wid;      // 4 | 64 nt per (mat,b,h): block-uniform upper bits
    if (job >= 4 * BSZ * HHEADS * (NSEQ / 16)) return;

    const int nt = job & 63;                    // row tile (16 rows)
    int t = job >> 6;
    const int h = t & 7;  t >>= 3;
    const int b = t & 7;
    const int mat = t >> 3;                     // 0..3 (block-uniform)

    const float* X; const float* W;
    if (mat == 0)      { X = query; W = qkern; }
    else if (mat == 1) { X = key;   W = kkern; }
    else if (mat == 2) { X = value; W = vkern; }
    else               { X = pos;   W = pkern; }
    const float* wbase = W + (size_t)h * DMODEL * DHEAD;

    const int half = lane >> 4, l15 = lane & 15;
    const int row  = nt * 16 + l15;
    const float* xrow = X + ((size_t)b * NSEQ + row) * DMODEL;

    CFrag acc[4];
    #pragma unroll
    for (int s = 0; s < 4; ++s)
        #pragma unroll
        for (int i = 0; i < 8; ++i) acc[s].f[i] = 0.0f;

    for (int k0 = 0; k0 < DMODEL; k0 += 32) {
        __syncthreads();
        fill_wtile(wt, wbase + (size_t)k0 * DHEAD, DHEAD, tid);
        // speculative prefetch of next k-slab row starts (global_prefetch_b8)
        if (tid < 32 && k0 + 32 < DMODEL)
            __builtin_prefetch(wbase + (size_t)(k0 + 32 + tid) * DHEAD, 0, 1);
        __syncthreads();

        // A fragment (16-bit A 16x32: K runs [8h..8h+7], [16+8h..16+8h+7])
        AFrag a;
        const float4* ap = (const float4*)(xrow + k0 + 8 * half);
        const float4 a0 = ap[0], a1 = ap[1], a2 = ap[4], a3 = ap[5];
        a.e[0]=(__bf16)a0.x; a.e[1]=(__bf16)a0.y; a.e[2]=(__bf16)a0.z; a.e[3]=(__bf16)a0.w;
        a.e[4]=(__bf16)a1.x; a.e[5]=(__bf16)a1.y; a.e[6]=(__bf16)a1.z; a.e[7]=(__bf16)a1.w;
        a.e[8]=(__bf16)a2.x; a.e[9]=(__bf16)a2.y; a.e[10]=(__bf16)a2.z; a.e[11]=(__bf16)a2.w;
        a.e[12]=(__bf16)a3.x; a.e[13]=(__bf16)a3.y; a.e[14]=(__bf16)a3.z; a.e[15]=(__bf16)a3.w;

        #pragma unroll
        for (int s = 0; s < 4; ++s) {
            // B fragment: lane N = l15 -> d = s*16+l15; K runs [8h..8h+7], [16+8h..16+8h+7]
            AFrag bf;
            const uint4* pb = (const uint4*)(&wt[s * 16 + l15][8 * half]);
            bf.q[0] = pb[0]; bf.q[1] = pb[2];
            acc[s].v = wmma_bf16(a.v, bf.v, acc[s].v);
        }
    }

    // store (C layout: VGPR r -> M = r + 8*half, N = lane&15)
    #pragma unroll
    for (int s = 0; s < 4; ++s) {
        const int dcol = s * 16 + l15;
        float bu = 0.f, bv = 0.f;
        if (mat == 0) { bu = bias_u[h * DHEAD + dcol]; bv = bias_v[h * DHEAD + dcol]; }
        #pragma unroll
        for (int r = 0; r < 8; ++r) {
            const int n = nt * 16 + r + 8 * half;
            const float av = acc[s].f[r];
            if (mat == 0) {
                const size_t idx = (((size_t)b * HHEADS + h) * NSEQ + n) * DHEAD + dcol;
                qu[idx] = bfBits(av + bu);
                qv[idx] = bfBits(av + bv);
            } else if (mat == 1) {
                kws[(((size_t)b * HHEADS + h) * MSEQ + n) * DHEAD + dcol] = bfBits(av);
            } else if (mat == 3) {
                pws[(((size_t)b * HHEADS + h) * MSEQ + n) * DHEAD + dcol] = bfBits(av);
            } else { // v transposed: [B,H,D,M]
                vt[(((size_t)b * HHEADS + h) * DHEAD + dcol) * MSEQ + n] = bfBits(av);
            }
        }
    }
}

// =====================================================================
// Kernel 2: fused attention for one (b, h, 32-row block).
// LDS: AC logits 32x1028 f32, BD 33x1028 f32 (row 32 = wrap row i+1),
// then BD region is reused as bf16 unnormalized softmax probabilities.
// =====================================================================
__global__ void __launch_bounds__(256) attn_kernel(
    const unsigned short* __restrict__ qu, const unsigned short* __restrict__ qv,
    const unsigned short* __restrict__ kws, const unsigned short* __restrict__ pws,
    const unsigned short* __restrict__ vt, unsigned short* __restrict__ ctx)
{
    extern __shared__ float smem[];
    float* lg = smem;                 // 32 x LGS
    float* bd = smem + 32 * LGS;      // 33 x LGS
    __shared__ float rsum[32];

    const int rblk = blockIdx.x & 31;
    const int h    = (blockIdx.x >> 5) & 7;
    const int b    = blockIdx.x >> 8;
    const int i0   = rblk * 32;

    const int tid  = threadIdx.x;
    const int wid  = tid >> 5;
    const int lane = tid & 31;
    const int half = lane >> 4, l15 = lane & 15;

    const size_t bh = (size_t)b * HHEADS + h;
    const unsigned short* qu_bh = qu  + bh * NSEQ * DHEAD;
    const unsigned short* qv_bh = qv  + bh * NSEQ * DHEAD;
    const unsigned short* k_bh  = kws + bh * MSEQ * DHEAD;
    const unsigned short* p_bh  = pws + bh * MSEQ * DHEAD;
    const unsigned short* v_bh  = vt  + bh * DHEAD * MSEQ;

    // ---- Step 1: AC (128 tiles) + BD (192 tiles incl. wrap row) via WMMA
    for (int t = wid; t < 320; t += 8) {
        int rt, ct; const unsigned short *asrc, *bsrc; bool isAC;
        if (t < 128) { isAC = true;  rt = t >> 6;          ct = t & 63;          asrc = qu_bh; bsrc = k_bh; }
        else         { isAC = false; rt = (t - 128) >> 6;  ct = (t - 128) & 63;  asrc = qv_bh; bsrc = p_bh; }
        const int grow0 = i0 + rt * 16;
        if (grow0 >= NSEQ) continue;   // last block: no wrap row exists (never read)

        CFrag acc;
        #pragma unroll
        for (int i = 0; i < 8; ++i) acc.f[i] = 0.0f;

        const unsigned short* arow = asrc + ((size_t)(grow0 + l15)) * DHEAD;
        const unsigned short* brow = bsrc + ((size_t)(ct * 16 + l15)) * DHEAD;
        #pragma unroll
        for (int k0 = 0; k0 < DHEAD; k0 += 32) {
            AFrag a, bf;
            const uint4* pa = (const uint4*)(arow + k0 + 8 * half);
            a.q[0] = pa[0]; a.q[1] = pa[2];
            const uint4* pb = (const uint4*)(brow + k0 + 8 * half);
            bf.q[0] = pb[0]; bf.q[1] = pb[2];
            acc.v = wmma_bf16(a.v, bf.v, acc.v);
        }
        #pragma unroll
        for (int r = 0; r < 8; ++r) {
            const int m   = r + 8 * half;
            const int col = ct * 16 + l15;
            if (isAC)          lg[(rt * 16 + m) * LGS + col] = acc.f[r];
            else if (rt < 2)   bd[(rt * 16 + m) * LGS + col] = acc.f[r];
            else if (m == 0)   bd[32 * LGS + col] = acc.f[r];  // wrap row i0+32
        }
    }
    __syncthreads();

    // ---- Step 2a: exact relative shift + combine + 1/sqrt(D)
    {
        const int r = tid >> 3;
        const int i = i0 + r;
        for (int c = tid & 7; c < MSEQ; c += 8) {
            float ac = lg[r * LGS + c];
            float sh;
            if (c <= i)          sh = bd[r * LGS + (MSEQ - 1 - i + c)];
            else if (c == i + 1) sh = 0.0f;
            else                 sh = bd[(r + 1) * LGS + (c - i - 2)];
            lg[r * LGS + c] = (ac + sh) * 0.125f;
        }
    }
    __syncthreads();

    // ---- Step 2b: row softmax (unnormalized probs as bf16 into bd region)
    unsigned short* attn = (unsigned short*)bd;
    #pragma unroll
    for (int rr = 0; rr < 4; ++rr) {
        const int r = wid * 4 + rr;
        float mx = -3.0e38f;
        for (int c = lane; c < MSEQ; c += 32) mx = fmaxf(mx, lg[r * LGS + c]);
        #pragma unroll
        for (int off = 16; off > 0; off >>= 1) mx = fmaxf(mx, __shfl_xor(mx, off, 32));
        float sum = 0.0f;
        for (int c = lane; c < MSEQ; c += 32) {
            float pv = __expf(lg[r * LGS + c] - mx);
            sum += pv;
            attn[r * ATTS + c] = bfBits(pv);
        }
        #pragma unroll
        for (int off = 16; off > 0; off >>= 1) sum += __shfl_xor(sum, off, 32);
        if (lane == 0) rsum[r] = 1.0f / sum;
    }
    __syncthreads();

    // ---- Step 3: ctx = softmax(logits) @ v ; fold 1/rowsum into C store
    {
        const int rt = wid >> 2;   // 0..1
        const int ct = wid & 3;    // 0..3  (16 cols each of D=64)
        CFrag acc;
        #pragma unroll
        for (int i = 0; i < 8; ++i) acc.f[i] = 0.0f;

        const unsigned short* vrow = v_bh + ((size_t)(ct * 16 + l15)) * MSEQ;
        for (int k0 = 0; k0 < MSEQ; k0 += 32) {
            AFrag a, bf;
            const uint4* pa = (const uint4*)(attn + (rt * 16 + l15) * ATTS + k0 + 8 * half);
            a.q[0] = pa[0]; a.q[1] = pa[2];
            const uint4* pb = (const uint4*)(vrow + k0 + 8 * half);
            bf.q[0] = pb[0]; bf.q[1] = pb[2];
            acc.v = wmma_bf16(a.v, bf.v, acc.v);
        }
        #pragma unroll
        for (int r = 0; r < 8; ++r) {
            const int m = rt * 16 + r + 8 * half;
            const int i = i0 + m;
            const float cv = acc.f[r] * rsum[m];
            ctx[((size_t)b * NSEQ + i) * (HHEADS * DHEAD) + h * DHEAD + ct * 16 + l15]
                = bfBits(cv);
        }
    }
}

// =====================================================================
// Kernel 3: out[8192,512] = ctx[8192,512](bf16) @ Wp[512,512](f32->bf16)
// Block = 4 waves sharing the same 64-col tile of Wp (rt fastest in job
// order) -> stage transposed bf16 weight tile in LDS per k-step.
// =====================================================================
__global__ void __launch_bounds__(128) outproj_kernel(
    const unsigned short* __restrict__ ctx, const float* __restrict__ proj,
    float* __restrict__ out)
{
    __shared__ unsigned short wt[DHEAD][WTS];   // wt[o_local][kk]

    const int tid  = threadIdx.x;
    const int wid  = tid >> 5;
    const int lane = tid & 31;
    const int job  = blockIdx.x * 4 + wid;
    if (job >= (BSZ * NSEQ / 16) * (DMODEL / 64)) return;

    const int ot = job >> 9;        // 64-col tile (block-uniform: 4 | 512)
    const int rt = job & 511;       // 16-row tile
    const int half = lane >> 4, l15 = lane & 15;

    const unsigned short* arow = ctx + ((size_t)rt * 16 + l15) * DMODEL;
    CFrag acc[4];
    #pragma unroll
    for (int s = 0; s < 4; ++s)
        #pragma unroll
        for (int i = 0; i < 8; ++i) acc[s].f[i] = 0.0f;

    for (int k0 = 0; k0 < DMODEL; k0 += 32) {
        __syncthreads();
        fill_wtile(wt, proj + (size_t)k0 * DMODEL + ot * 64, DMODEL, tid);
        if (tid < 32 && k0 + 32 < DMODEL)
            __builtin_prefetch(proj + (size_t)(k0 + 32 + tid) * DMODEL + ot * 64, 0, 1);
        __syncthreads();

        AFrag a;
        const uint4* pa = (const uint4*)(arow + k0 + 8 * half);
        a.q[0] = pa[0]; a.q[1] = pa[2];
        #pragma unroll
        for (int s = 0; s < 4; ++s) {
            AFrag bf;
            const uint4* pb = (const uint4*)(&wt[s * 16 + l15][8 * half]);
            bf.q[0] = pb[0]; bf.q[1] = pb[2];
            acc[s].v = wmma_bf16(a.v, bf.v, acc[s].v);
        }
    }
    #pragma unroll
    for (int s = 0; s < 4; ++s)
        #pragma unroll
        for (int r = 0; r < 8; ++r) {
            const size_t row = (size_t)rt * 16 + r + 8 * half;
            out[row * DMODEL + ot * 64 + s * 16 + l15] = acc[s].f[r];
        }
}

// =====================================================================
extern "C" void kernel_launch(void* const* d_in, const int* in_sizes, int n_in,
                              void* d_out, int out_size, void* d_ws, size_t ws_size,
                              hipStream_t stream)
{
    const float* query  = (const float*)d_in[0];
    const float* key    = (const float*)d_in[1];
    const float* value  = (const float*)d_in[2];
    const float* pos    = (const float*)d_in[3];
    const float* qkern  = (const float*)d_in[4];
    const float* kkern  = (const float*)d_in[5];
    const float* vkern  = (const float*)d_in[6];
    const float* pkern  = (const float*)d_in[7];
    const float* projk  = (const float*)d_in[8];
    const float* bias_u = (const float*)d_in[9];
    const float* bias_v = (const float*)d_in[10];
    float* out = (float*)d_out;

    // bf16 workspace: q_u, q_v, k, p, v^T, ctx  (6 x 8 MiB = 48 MiB)
    unsigned short* ws = (unsigned short*)d_ws;
    const size_t SZ = (size_t)BSZ * HHEADS * NSEQ * DHEAD;
    unsigned short* qu  = ws;
    unsigned short* qv  = ws + SZ;
    unsigned short* kws = ws + 2 * SZ;
    unsigned short* pws = ws + 3 * SZ;
    unsigned short* vt  = ws + 4 * SZ;
    unsigned short* ctx = ws + 5 * SZ;

    // 1) projections: 4 mats * 8 b * 8 h * 64 row-tiles = 16384 waves
    proj_kernel<<<4096, 128, 0, stream>>>(query, key, value, pos,
                                          qkern, kkern, vkern, pkern,
                                          bias_u, bias_v,
                                          qu, qv, kws, pws, vt);

    // 2) fused attention: B*H*(N/32) = 2048 blocks, 267,280 B dynamic LDS
    const size_t shbytes = (size_t)(32 * LGS + 33 * LGS) * sizeof(float);
    attn_kernel<<<2048, 256, shbytes, stream>>>(qu, qv, kws, pws, vt, ctx);

    // 3) output projection: 512 row-tiles * 8 col-tiles = 4096 waves
    outproj_kernel<<<1024, 128, 0, stream>>>(ctx, projk, out);
}
